// MultiHeadAttentionLayer_42399917146346
// MI455X (gfx1250) — matrix operand order repaired and checked by reading, not compiled, not run on hardware
//
#include <hip/hip_runtime.h>
#include <hip/hip_bf16.h>
#include <math.h>

// Problem constants (fixed by the reference)
#define BATCH   4
#define SEQ     2048
#define HID     1024
#define NHEAD   16
#define DHEAD   64
#define NEGINF  (-1.0e9f)

typedef __attribute__((ext_vector_type(16))) __bf16 v16bf;
typedef __attribute__((ext_vector_type(8)))  __bf16 v8bf;
typedef __attribute__((ext_vector_type(8)))  float  v8f;

__device__ __forceinline__ v16bf combine16(v8bf lo, v8bf hi) {
    v16bf r;
#pragma unroll
    for (int i = 0; i < 8; ++i) { r[i] = lo[i]; r[8 + i] = hi[i]; }
    return r;
}

// Epilogue modes
#define EPI_BIAS   0   // C = acc + bias           (row-major)
#define EPI_BHSD   1   // C = acc + bias, scattered to [B,H,S,D] head-major
#define EPI_MASK   2   // C = acc*scale + mask*NEG (row-major, batched)
#define EPI_PLAIN  3   // C = acc                  (row-major, batched)

// ---------------------------------------------------------------------------
// Batched GEMM: C[z] = A[z] (MxK) @ B[z] (KxN, or NxK if TRANSB) [+ bias]
// f32 in/out, bf16 WMMA 16x16x32 compute, f32 accumulate.
// Tile 128x128, BK=32, 8 waves; each wave 32x64 = 2x4 WMMA accumulators.
// Requirements: M % 128 == 0, K % 32 == 0. N % 128 == 0 unless GUARD_N.
// ---------------------------------------------------------------------------
template<int TRANSB, int EPI, bool GUARD_N>
__global__ __launch_bounds__(256)
void mha_gemm_bf16_wmma(const float* __restrict__ A, long lda, long strideAz,
                        const float* __restrict__ B, long ldb, long strideBz,
                        const float* __restrict__ bias,
                        float* __restrict__ C, long ldc,
                        long c_div, long c_outer, long c_inner,
                        int M, int N, int K,
                        const int* __restrict__ mask, long mask_div,
                        float scale, float mneg)
{
    __shared__ __bf16 As[128][40];   // MxK tile (padded)
    __shared__ __bf16 Bst[128][40];  // NxK tile (padded; B staged transposed)

    const int tid  = threadIdx.x;
    const int lane = tid & 31;
    const int wave = tid >> 5;
    const int wm   = wave & 3;   // 4 waves along M  -> 32 rows each
    const int wn   = wave >> 2;  // 2 waves along N  -> 64 cols each
    const int lr   = lane & 15;
    const int lh   = lane >> 4;

    const int z     = blockIdx.z;
    const int tileM = blockIdx.y * 128;
    const int tileN = blockIdx.x * 128;

    const float* Az = A + (long)z * strideAz;
    const float* Bz = B + (long)z * strideBz;

    v8f acc[2][4];
#pragma unroll
    for (int i = 0; i < 2; ++i)
#pragma unroll
        for (int j = 0; j < 4; ++j) acc[i][j] = (v8f)(0.0f);

    for (int k0 = 0; k0 < K; k0 += 32) {
        // ---- stage A tile (128x32) -> bf16 row-major; no guards (M%128==0) ----
#pragma unroll
        for (int i = 0; i < 4; ++i) {
            int u   = tid + i * 256;        // 0..1023 float4 units
            int row = u >> 3;               // 0..127
            int c4  = (u & 7) << 2;         // 0,4,...,28
            float4 f = *reinterpret_cast<const float4*>(
                &Az[(long)(tileM + row) * lda + k0 + c4]);
            As[row][c4 + 0] = (__bf16)f.x;
            As[row][c4 + 1] = (__bf16)f.y;
            As[row][c4 + 2] = (__bf16)f.z;
            As[row][c4 + 3] = (__bf16)f.w;
        }
        // ---- stage B tile -> bf16 transposed into Bst[N][K] ----
        if (TRANSB) {
            // B is [N x K] row-major (e.g. K-heads for Q@K^T): direct copy
#pragma unroll
            for (int i = 0; i < 4; ++i) {
                int u  = tid + i * 256;
                int n  = u >> 3;            // 0..127
                int c4 = (u & 7) << 2;      // 0..28
                long gn = (long)tileN + n;
                // branchless guard: clamp address, zero value
                long gs = GUARD_N ? (gn < N ? gn : 0) : gn;
                float4 f = *reinterpret_cast<const float4*>(
                    &Bz[gs * ldb + k0 + c4]);
                if (GUARD_N && gn >= N) f = make_float4(0.f, 0.f, 0.f, 0.f);
                Bst[n][c4 + 0] = (__bf16)f.x;
                Bst[n][c4 + 1] = (__bf16)f.y;
                Bst[n][c4 + 2] = (__bf16)f.z;
                Bst[n][c4 + 3] = (__bf16)f.w;
            }
        } else {
            // B is [K x N] row-major: transpose while staging
#pragma unroll
            for (int i = 0; i < 4; ++i) {
                int u  = tid + i * 256;
                int kk = u >> 5;            // 0..31
                int n4 = (u & 31) << 2;     // 0..124
                long gn = (long)tileN + n4;
                long gs = GUARD_N ? (gn < N ? gn : 0) : gn;
                float4 f = *reinterpret_cast<const float4*>(
                    &Bz[(long)(k0 + kk) * ldb + gs]);
                if (GUARD_N && gn >= N) f = make_float4(0.f, 0.f, 0.f, 0.f);
                Bst[n4 + 0][kk] = (__bf16)f.x;
                Bst[n4 + 1][kk] = (__bf16)f.y;
                Bst[n4 + 2][kk] = (__bf16)f.z;
                Bst[n4 + 3][kk] = (__bf16)f.w;
            }
        }
        // prefetch next K-tile of A into cache (global_prefetch_b8)
        if (k0 + 32 < K)
            __builtin_prefetch(&Az[(long)(tileM + (tid & 127)) * lda + k0 + 32], 0, 1);
        __syncthreads();

        // ---- fragments (ISA 7.12.2 layouts) + 8 WMMAs per wave ----
        v16bf afrag[2];
#pragma unroll
        for (int im = 0; im < 2; ++im) {
            int r = wm * 32 + im * 16 + lr;
            v8bf lo = *reinterpret_cast<const v8bf*>(&As[r][lh * 8]);
            v8bf hi = *reinterpret_cast<const v8bf*>(&As[r][16 + lh * 8]);
            afrag[im] = combine16(lo, hi);
        }
        v16bf bfrag[4];
#pragma unroll
        for (int jn = 0; jn < 4; ++jn) {
            int c = wn * 64 + jn * 16 + lr;
            v8bf lo = *reinterpret_cast<const v8bf*>(&Bst[c][lh * 8]);
            v8bf hi = *reinterpret_cast<const v8bf*>(&Bst[c][16 + lh * 8]);
            bfrag[jn] = combine16(lo, hi);
        }
#pragma unroll
        for (int im = 0; im < 2; ++im)
#pragma unroll
            for (int jn = 0; jn < 4; ++jn)
                acc[im][jn] = __builtin_amdgcn_wmma_f32_16x16x32_bf16(
                    false, afrag[im], false, bfrag[jn],
                    (short)0, acc[im][jn], false, false);
        __syncthreads();
    }

    // ---- epilogue (straight-line per template mode) ----
    const long zco = (z / c_div) * c_outer + (z % c_div) * c_inner;
    const long maskbase = (EPI == EPI_MASK)
                        ? (z / mask_div) * ((long)M * N) : 0;
#pragma unroll
    for (int im = 0; im < 2; ++im) {
#pragma unroll
        for (int jn = 0; jn < 4; ++jn) {
            int gnc = tileN + wn * 64 + jn * 16 + lr;
            if (GUARD_N && gnc >= N) continue;           // wave-uniform skip
            float bb = (EPI == EPI_BIAS || EPI == EPI_BHSD) ? bias[gnc] : 0.0f;
#pragma unroll
            for (int r = 0; r < 8; ++r) {
                int gm = tileM + wm * 32 + im * 16 + r + (lh << 3);
                float v = acc[im][jn][r];
                if (EPI == EPI_MASK) {
                    v = v * scale
                      + (float)mask[maskbase + (long)gm * N + gnc] * mneg;
                } else if (EPI == EPI_BIAS || EPI == EPI_BHSD) {
                    v += bb;
                }
                if (EPI == EPI_BHSD) {
                    int b = gm >> 11, s = gm & (SEQ - 1);
                    int h = gnc >> 6, d = gnc & (DHEAD - 1);
                    C[(((long)(b * NHEAD + h) * SEQ + s) * DHEAD) + d] = v;
                } else {
                    C[zco + (long)gm * ldc + gnc] = v;
                }
            }
        }
    }
}

// ---------------------------------------------------------------------------
// In-place row softmax over rows of length SEQ (2048). One block per row.
// ---------------------------------------------------------------------------
__global__ __launch_bounds__(256)
void mha_softmax_rows(float* __restrict__ P)
{
    __shared__ float red[256];
    const int t = threadIdx.x;
    float* p = P + (long)blockIdx.x * SEQ;

    float vals[8];
    float lmax = -INFINITY;
#pragma unroll
    for (int j = 0; j < 8; ++j) {
        vals[j] = p[t + j * 256];
        lmax = fmaxf(lmax, vals[j]);
    }
    red[t] = lmax;
    __syncthreads();
    for (int s = 128; s > 0; s >>= 1) {
        if (t < s) red[t] = fmaxf(red[t], red[t + s]);
        __syncthreads();
    }
    const float m = red[0];
    __syncthreads();

    float lsum = 0.0f;
#pragma unroll
    for (int j = 0; j < 8; ++j) {
        vals[j] = __expf(vals[j] - m);
        lsum += vals[j];
    }
    red[t] = lsum;
    __syncthreads();
    for (int s = 128; s > 0; s >>= 1) {
        if (t < s) red[t] += red[t + s];
        __syncthreads();
    }
    const float inv = 1.0f / red[0];
#pragma unroll
    for (int j = 0; j < 8; ++j)
        p[t + j * 256] = vals[j] * inv;
}

// ---------------------------------------------------------------------------
extern "C" void kernel_launch(void* const* d_in, const int* in_sizes, int n_in,
                              void* d_out, int out_size, void* d_ws, size_t ws_size,
                              hipStream_t stream)
{
    const float* Q    = (const float*)d_in[0];
    const float* K    = (const float*)d_in[1];
    const float* V    = (const float*)d_in[2];
    const int*   msk  = (const int*)  d_in[3];
    const float* Wq   = (const float*)d_in[4];
    const float* bq   = (const float*)d_in[5];
    const float* Wk   = (const float*)d_in[6];
    const float* bk   = (const float*)d_in[7];
    const float* Wv   = (const float*)d_in[8];
    const float* bv   = (const float*)d_in[9];
    const float* Wo   = (const float*)d_in[10];
    const float* bo   = (const float*)d_in[11];

    const long MROWS = (long)BATCH * SEQ;          // 8192
    float* out   = (float*)d_out;                  // [B,S,HID]
    float* probs = out + MROWS * HID;              // [B,H,S,S]

    float* qh  = (float*)d_ws;                     // [B,H,S,D] head-major
    float* kh  = qh + MROWS * HID;
    float* vh  = kh + MROWS * HID;
    float* ctx = vh + MROWS * HID;                 // [B,S,H*D] concat layout

    dim3 blk(256);
    const float scale = 0.125f;                    // 1/sqrt(64)

    // 1-3. QKV projections -> head-major [B,H,S,D]
    dim3 gproj(HID / 128, MROWS / 128, 1);         // 8 x 64
    mha_gemm_bf16_wmma<0, EPI_BHSD, false><<<gproj, blk, 0, stream>>>(
        Q, HID, 0, Wq, HID, 0, bq, qh, HID, 1, 0, 0,
        (int)MROWS, HID, HID, nullptr, 1, 1.0f, 0.0f);
    mha_gemm_bf16_wmma<0, EPI_BHSD, false><<<gproj, blk, 0, stream>>>(
        K, HID, 0, Wk, HID, 0, bk, kh, HID, 1, 0, 0,
        (int)MROWS, HID, HID, nullptr, 1, 1.0f, 0.0f);
    mha_gemm_bf16_wmma<0, EPI_BHSD, false><<<gproj, blk, 0, stream>>>(
        V, HID, 0, Wv, HID, 0, bv, vh, HID, 1, 0, 0,
        (int)MROWS, HID, HID, nullptr, 1, 1.0f, 0.0f);

    // 4. logits = scale*(q @ k^T) + mask*NEG, one z per (b,h), into probs
    dim3 glog(SEQ / 128, SEQ / 128, BATCH * NHEAD); // 16 x 16 x 64
    mha_gemm_bf16_wmma<1, EPI_MASK, false><<<glog, blk, 0, stream>>>(
        qh, DHEAD, (long)SEQ * DHEAD,
        kh, DHEAD, (long)SEQ * DHEAD,
        nullptr, probs, SEQ,
        1, (long)SEQ * SEQ, 0,
        SEQ, SEQ, DHEAD,
        msk, NHEAD, scale, NEGINF);

    // 5. row softmax in place on probs
    mha_softmax_rows<<<BATCH * NHEAD * SEQ, blk, 0, stream>>>(probs);

    // 6. ctx = probs @ v  -> written straight into concat layout [B,S,H*D]
    dim3 gctx(1, SEQ / 128, BATCH * NHEAD);
    mha_gemm_bf16_wmma<0, EPI_PLAIN, true><<<gctx, blk, 0, stream>>>(
        probs, SEQ, (long)SEQ * SEQ,
        vh, DHEAD, (long)SEQ * DHEAD,
        nullptr, ctx, HID,
        NHEAD, (long)SEQ * HID, DHEAD,
        SEQ, DHEAD, SEQ,
        nullptr, 1, 1.0f, 0.0f);

    // 7. out = ctx @ Wo + bo
    mha_gemm_bf16_wmma<0, EPI_BIAS, false><<<gproj, blk, 0, stream>>>(
        ctx, HID, 0, Wo, HID, 0, bo, out, HID, 1, 0, 0,
        (int)MROWS, HID, HID, nullptr, 1, 1.0f, 0.0f);
}
